// SpatioTemporalGCN_36197984371321
// MI455X (gfx1250) — matrix operand order repaired and checked by reading, not compiled, not run on hardware
//
#include <hip/hip_runtime.h>
#include <hip/hip_bf16.h>

typedef __bf16 bf16_t;
typedef __attribute__((ext_vector_type(16))) __bf16    v16bf;
typedef __attribute__((ext_vector_type(8)))  float     v8f;
typedef __attribute__((ext_vector_type(4)))  unsigned int u32x4;

#define B_   64
#define T_   24
#define TP_  (T_ + 2)          // padded timesteps for conv (zero rows at 0 and 25)
#define N_   300
#define F_   16
#define D_   64
#define C1_  128
#define C2_  128
#define H_   12
#define ND_  (N_*D_)           // 19200
#define CONVK_ (3*ND_)         // 57600
#define NKT_   (ND_/32)        // 600 k-tiles per conv tap
#define MLPIN_ (ND_ + C2_)     // 19328
#define MH1_ 512
#define MH2_ 256
#define OUT_ (N_*H_)           // 3600

union H8 { u32x4 u; bf16_t h[8]; };

__device__ __forceinline__ H8 h8_zero() {
  H8 r; r.u = (u32x4){0u, 0u, 0u, 0u}; return r;
}
__device__ __forceinline__ H8 h8_load(const bf16_t* p) {
  H8 r; r.u = *(const u32x4*)p; return r;
}
__device__ __forceinline__ H8 h8_cvt(const float* p) {
  H8 r;
#pragma unroll
  for (int i = 0; i < 8; ++i) r.h[i] = (bf16_t)p[i];
  return r;
}
__device__ __forceinline__ v16bf mk_frag(H8 lo, H8 hi) {
  v16bf a;
#pragma unroll
  for (int i = 0; i < 8; ++i) { a[i] = lo.h[i]; a[8 + i] = hi.h[i]; }
  return a;
}
__device__ __forceinline__ v8f wmma_bf16(v16bf a, v16bf b, v8f c) {
  return __builtin_amdgcn_wmma_f32_16x16x32_bf16(false, a, false, b,
                                                 (short)0, c, false, false);
}
// Async copy of 16 bytes from global to LDS (per lane), tracked by ASYNCcnt.
__device__ __forceinline__ void async_cp16(unsigned lds_off, const bf16_t* g) {
  asm volatile("global_load_async_to_lds_b128 %0, %1, off"
               :: "v"(lds_off), "v"(g) : "memory");
}

// ---------------------------------------------------------------------------
// prep kernels: f32 -> bf16, transposed / padded layouts
// ---------------------------------------------------------------------------
__global__ void k_prep_adj(const float* __restrict__ adj, bf16_t* __restrict__ adjp) {
  int i = blockIdx.x * 256 + threadIdx.x;           // [304][320]
  if (i >= 304 * 320) return;
  int m = i / 320, k = i % 320;
  adjp[i] = (m < N_ && k < N_) ? (bf16_t)adj[m * N_ + k] : (bf16_t)0.0f;
}

__global__ void k_prep_gcnw(const float* __restrict__ w1, const float* __restrict__ w2,
                            bf16_t* __restrict__ wt1, bf16_t* __restrict__ wt2) {
  int i = blockIdx.x * 256 + threadIdx.x;
  if (i < 64 * 32) {                                // wt1[d][k], k padded 16->32
    int d = i / 32, k = i % 32;
    wt1[i] = (k < F_) ? (bf16_t)w1[k * D_ + d] : (bf16_t)0.0f;
  }
  if (i < 64 * 64) {                                // wt2[d][k]
    int d = i / 64, k = i % 64;
    wt2[i] = (bf16_t)w2[k * D_ + d];
  }
}

__global__ void k_prep_convw(const float* __restrict__ w1, bf16_t* __restrict__ w1t) {
  int i = blockIdx.x * 256 + threadIdx.x;           // c*57600 + k*19200 + cin
  if (i >= C1_ * CONVK_) return;
  int c = i / CONVK_;
  int rem = i % CONVK_;
  int k = rem / ND_, cin = rem % ND_;
  w1t[i] = (bf16_t)w1[((size_t)c * ND_ + cin) * 3 + k];
}

__global__ void k_prep_mlpw(const float* __restrict__ w1, bf16_t* __restrict__ w1t) {
  int i = blockIdx.x * 256 + threadIdx.x;           // n*19328 + k
  if (i >= MLPIN_ * MH1_) return;
  int n = i / MLPIN_, k = i % MLPIN_;
  w1t[i] = (bf16_t)w1[(size_t)k * MH1_ + n];
}

// zero the t=0 and t=TP_-1 boundary rows of the padded h2 buffer
__global__ void k_zero_pad(bf16_t* __restrict__ h2p) {
  int i = blockIdx.x * 256 + threadIdx.x;
  if (i >= B_ * 2 * ND_) return;
  int cin = i % ND_;
  int r = (i / ND_) % 2;
  int b = i / (2 * ND_);
  h2p[((size_t)b * TP_ + r * (TP_ - 1)) * ND_ + cin] = (bf16_t)0.0f;
}

// ---------------------------------------------------------------------------
// GCN layer, fused per (b,t): support = in @ W into LDS (transposed, K padded
// 300->320), then h = relu(adj @ support + bias). Output row mapping is
// parameterized so layer 2 writes straight into the T-padded conv buffer.
// ---------------------------------------------------------------------------
template <int KIN, bool IN_F32>
__global__ __launch_bounds__(256) void k_gcn(const void* __restrict__ inp,
                                             const bf16_t* __restrict__ wt,    // [64][KPAD]
                                             const bf16_t* __restrict__ adjp,  // [304][320]
                                             const float* __restrict__ bias,   // [64]
                                             bf16_t* __restrict__ out,
                                             int trows, int toff)
{
  constexpr int KPAD = (KIN == 16) ? 32 : 64;
  constexpr int KT1  = KPAD / 32;
  __shared__ __align__(16) bf16_t sT[64][320];      // support^T, 40 KB

  const int bt   = blockIdx.x;
  const int bb   = bt / T_;
  const int tt   = bt % T_;
  const size_t orow = ((size_t)bb * trows + tt + toff) * N_;
  const int lane = threadIdx.x & 31;
  const int wave = threadIdx.x >> 5;
  const int half = lane >> 4;
  const int lm   = lane & 15;

  for (int idx = threadIdx.x; idx < 64 * 16; idx += 256) {
    int d = idx >> 4, n = 304 + (idx & 15);
    sT[d][n] = (bf16_t)0.0f;
  }

  for (int tile = wave; tile < 19 * 4; tile += 8) {
    const int mt = tile >> 2, nt = tile & 3;
    const int n  = mt * 16 + lm;
    const int d  = nt * 16 + lm;
    v8f acc = {};
#pragma unroll
    for (int kt = 0; kt < KT1; ++kt) {
      H8 alo, ahi;
      if (n < N_) {
        if (IN_F32) {
          const float* xp = (const float*)inp + ((size_t)bt * N_ + n) * F_;
          alo = h8_cvt(xp + half * 8);              // K = half*8+e (< 16)
          ahi = h8_zero();                           // K >= 16 zero-padded
        } else {
          const bf16_t* hp = (const bf16_t*)inp + ((size_t)bt * N_ + n) * D_;
          alo = h8_load(hp + kt * 32 + half * 8);
          ahi = h8_load(hp + kt * 32 + 16 + half * 8);
        }
      } else { alo = h8_zero(); ahi = h8_zero(); }
      const bf16_t* wp = wt + (size_t)d * KPAD + kt * 32 + half * 8;
      H8 blo = h8_load(wp);
      H8 bhi = h8_load(wp + 16);
      acc = wmma_bf16(mk_frag(alo, ahi), mk_frag(blo, bhi), acc);
    }
    H8 s;
#pragma unroll
    for (int v = 0; v < 8; ++v) s.h[v] = (bf16_t)acc[v];
    const int nbase = mt * 16 + half * 8;
    *(u32x4*)&sT[d][nbase] = s.u;
  }
  __syncthreads();

  for (int tile = wave; tile < 19 * 4; tile += 8) {
    const int mt = tile >> 2, nt = tile & 3;
    const int m  = mt * 16 + lm;
    const int d  = nt * 16 + lm;
    v8f acc = {};
    for (int kt = 0; kt < 10; ++kt) {               // K = 320
      const bf16_t* ap = adjp + (size_t)m * 320 + kt * 32 + half * 8;
      H8 alo = h8_load(ap);
      H8 ahi = h8_load(ap + 16);
      const bf16_t* bp = &sT[d][kt * 32 + half * 8];
      H8 blo = h8_load(bp);
      H8 bhi = h8_load(bp + 16);
      acc = wmma_bf16(mk_frag(alo, ahi), mk_frag(blo, bhi), acc);
    }
    const float bv = bias[d];
#pragma unroll
    for (int v = 0; v < 8; ++v) {
      const int mo = mt * 16 + half * 8 + v;
      if (mo < N_) {
        float r = acc[v] + bv;
        out[(orow + mo) * D_ + d] = (bf16_t)fmaxf(r, 0.0f);
      }
    }
  }
}

// ---------------------------------------------------------------------------
// conv1 as im2col GEMM: M = B*T = 1536, N = C1 = 128, K = 57600.
// A streams contiguous bf16 rows of T-padded h2 (no division, no predication).
// B (weights) double-buffered into LDS with global_load_async_to_lds_b128.
// ---------------------------------------------------------------------------
__global__ __launch_bounds__(256) void k_conv1(const bf16_t* __restrict__ h2p,  // (B,26,19200)
                                               const bf16_t* __restrict__ w1t,  // [128][57600]
                                               const float* __restrict__ b1,
                                               float* __restrict__ y1)          // (B,128,24)
{
  __shared__ __align__(16) bf16_t sB[8][2][512];    // [wave][buf][16 rows x 32 k] = 16 KB
  const int lane = threadIdx.x & 31;
  const int wave = threadIdx.x >> 5;
  const int half = lane >> 4;
  const int lm   = lane & 15;
  const int mt = blockIdx.x;                         // 0..95
  const int nt = wave;                               // 0..7
  const int r  = mt * 16 + lm;                       // row = (b,t)
  const int b  = r / T_;
  const int t  = r % T_;
  const int c  = nt * 16 + lm;

  // async staging lane mapping: lane -> (row = l>>2 of 16, seg = l&3 of 4)
  const int row0 = lane >> 2;                        // 0..7
  const int seg  = lane & 3;                         // 0..3
  const bf16_t* gB0 = w1t + ((size_t)(nt * 16 + row0)) * CONVK_ + seg * 8;
  const bf16_t* gB1 = gB0 + (size_t)8 * CONVK_;
  const unsigned ldsBase =
      (unsigned)(unsigned long long)(void*)&sB[wave][0][0] + row0 * 64 + seg * 16;

  auto issue = [&](int idx) {
    unsigned l = ldsBase + (unsigned)(idx & 1) * 1024u;
    async_cp16(l,        gB0 + (size_t)idx * 32);    // rows 0..7 of tile
    async_cp16(l + 512u, gB1 + (size_t)idx * 32);    // rows 8..15 of tile
  };

  issue(0);
  const bf16_t* arow = h2p + ((size_t)b * TP_ + t) * ND_;  // tap kc -> + kc*ND_
  v8f acc = {};
  int idx = 0;
  for (int kc = 0; kc < 3; ++kc) {
    const bf16_t* ar = arow + (size_t)kc * ND_;
    for (int kt = 0; kt < NKT_; ++kt, ++idx) {
      if (idx + 1 < 3 * NKT_) {
        issue(idx + 1);
        asm volatile("s_wait_asynccnt 0x2" ::: "memory");   // current tile landed
      } else {
        asm volatile("s_wait_asynccnt 0x0" ::: "memory");   // last tile: drain
      }
      H8 alo = h8_load(ar + kt * 32 + half * 8);
      H8 ahi = h8_load(ar + kt * 32 + 16 + half * 8);
      __builtin_prefetch(ar + kt * 32 + 1024, 0, 1);
      const bf16_t* lb = &sB[wave][idx & 1][0];
      H8 blo = h8_load(lb + lm * 32 + half * 8);
      H8 bhi = h8_load(lb + lm * 32 + half * 8 + 16);
      acc = wmma_bf16(mk_frag(alo, ahi), mk_frag(blo, bhi), acc);
    }
  }
  const float bv = b1[c];
#pragma unroll
  for (int v = 0; v < 8; ++v) {
    const int ro = mt * 16 + half * 8 + v;
    const int bo = ro / T_, to = ro % T_;
    y1[((size_t)bo * C1_ + c) * T_ + to] = fmaxf(acc[v] + bv, 0.0f);
  }
}

// ---------------------------------------------------------------------------
// conv2 (small, VALU f32), temporal mean pool, MLP head
// ---------------------------------------------------------------------------
__global__ void k_conv2(const float* __restrict__ y1, const float* __restrict__ w2,
                        const float* __restrict__ b2, float* __restrict__ y2)
{
  int i = blockIdx.x * 256 + threadIdx.x;           // (b*C2 + c)*T + t
  if (i >= B_ * C2_ * T_) return;
  int t = i % T_;
  int c = (i / T_) % C2_;
  int b = i / (T_ * C2_);
  float acc = b2[c];
  for (int k = 0; k < 3; ++k) {
    int tt = t + k - 1;
    if (tt < 0 || tt >= T_) continue;
    const float* yp = y1 + (size_t)b * C1_ * T_ + tt;
    const float* wp = w2 + (size_t)c * C1_ * 3 + k;
    for (int ci = 0; ci < C1_; ++ci)
      acc += wp[ci * 3] * yp[(size_t)ci * T_];
  }
  y2[i] = fmaxf(acc, 0.0f);
}

__global__ void k_pool(const float* __restrict__ y2, bf16_t* __restrict__ pooled) {
  int i = blockIdx.x * 256 + threadIdx.x;           // b*128 + c
  if (i >= B_ * C2_) return;
  const float* p = y2 + (size_t)i * T_;
  float s = 0.0f;
  for (int t = 0; t < T_; ++t) s += p[t];
  pooled[i] = (bf16_t)(s * (1.0f / T_));
}

// MLP1: M=64, N=512, K=19328 (=604*32). A row = [h2[b,T-1,:] | pooled[b,:]].
__global__ __launch_bounds__(256) void k_mlp1(const bf16_t* __restrict__ h2p,
                                              const bf16_t* __restrict__ pooled,
                                              const bf16_t* __restrict__ w1t,   // [512][19328]
                                              const float* __restrict__ b1,
                                              float* __restrict__ z1)           // [64][512]
{
  const int lane = threadIdx.x & 31;
  const int wave = threadIdx.x >> 5;
  const int half = lane >> 4;
  const int lm   = lane & 15;
  const int gw = blockIdx.x * 8 + wave;             // 0..127 wave-tiles
  const int mt = gw >> 5;                           // 0..3
  const int nt = gw & 31;                           // 0..31
  const int bm = mt * 16 + lm;                      // batch row (< 64)
  const int n  = nt * 16 + lm;
  const bf16_t* arow = h2p + ((size_t)bm * TP_ + T_) * ND_;  // padded row of t = T-1
  const bf16_t* prow = pooled + (size_t)bm * C2_;
  const bf16_t* bp0  = w1t + (size_t)n * MLPIN_;
  v8f acc = {};
  for (int kt = 0; kt < MLPIN_ / 32; ++kt) {        // 604 iterations
    const int kb = kt * 32;
    H8 alo, ahi;
    { int kk = kb + half * 8;
      alo = (kk < ND_) ? h8_load(arow + kk) : h8_load(prow + (kk - ND_)); }
    { int kk = kb + 16 + half * 8;
      ahi = (kk < ND_) ? h8_load(arow + kk) : h8_load(prow + (kk - ND_)); }
    H8 blo = h8_load(bp0 + kb + half * 8);
    H8 bhi = h8_load(bp0 + kb + half * 8 + 16);
    acc = wmma_bf16(mk_frag(alo, ahi), mk_frag(blo, bhi), acc);
  }
  const float bv = b1[n];
#pragma unroll
  for (int v = 0; v < 8; ++v) {
    const int bo = mt * 16 + half * 8 + v;
    z1[(size_t)bo * MH1_ + n] = fmaxf(acc[v] + bv, 0.0f);
  }
}

__global__ void k_mlp2(const float* __restrict__ z1, const float* __restrict__ w2,
                       const float* __restrict__ b2, float* __restrict__ z2) {
  int i = blockIdx.x * 256 + threadIdx.x;           // b*256 + j
  if (i >= B_ * MH2_) return;
  int j = i % MH2_, b = i / MH2_;
  const float* zr = z1 + (size_t)b * MH1_;
  float acc = b2[j];
  for (int k = 0; k < MH1_; ++k) acc += zr[k] * w2[(size_t)k * MH2_ + j];
  z2[i] = fmaxf(acc, 0.0f);
}

__global__ void k_mlp3(const float* __restrict__ z2, const float* __restrict__ w3,
                       const float* __restrict__ b3, float* __restrict__ out) {
  int i = blockIdx.x * 256 + threadIdx.x;           // b*3600 + o
  if (i >= B_ * OUT_) return;
  int o = i % OUT_, b = i / OUT_;
  const float* zr = z2 + (size_t)b * MH2_;
  float acc = b3[o];
  for (int k = 0; k < MH2_; ++k) acc += zr[k] * w3[(size_t)k * OUT_ + o];
  out[i] = acc;
}

// ---------------------------------------------------------------------------
// Host launcher. Workspace (bytes, 256-aligned), total ~159.4 MB:
//   adjp @0 194,560 | wt1 @194,560 4,096 | wt2 @198,656 8,192
//   h1   @206,848     58,982,400 bf16 (B*T,300,64)
//   h2p  @59,189,248  63,897,600 bf16 (B,26,19200), rows 0/25 zeroed
//   cw1t @123,086,848 14,745,600 | mw1t @137,832,448 19,791,872
//   y1 @157,624,320 786,432 | y2 @158,410,752 786,432
//   pooled @159,197,184 16,384 | z1 @159,213,568 131,072 | z2 @159,344,640 65,536
// ---------------------------------------------------------------------------
extern "C" void kernel_launch(void* const* d_in, const int* in_sizes, int n_in,
                              void* d_out, int out_size, void* d_ws, size_t ws_size,
                              hipStream_t stream)
{
  (void)in_sizes; (void)n_in; (void)out_size; (void)ws_size;
  const float* x   = (const float*)d_in[0];
  const float* adj = (const float*)d_in[1];
  const float* gw1 = (const float*)d_in[2];
  const float* gb1 = (const float*)d_in[3];
  const float* gw2 = (const float*)d_in[4];
  const float* gb2 = (const float*)d_in[5];
  const float* cw1 = (const float*)d_in[6];
  const float* cb1 = (const float*)d_in[7];
  const float* cw2 = (const float*)d_in[8];
  const float* cb2 = (const float*)d_in[9];
  const float* mw1 = (const float*)d_in[10];
  const float* mb1 = (const float*)d_in[11];
  const float* mw2 = (const float*)d_in[12];
  const float* mb2 = (const float*)d_in[13];
  const float* mw3 = (const float*)d_in[14];
  const float* mb3 = (const float*)d_in[15];
  float* out = (float*)d_out;

  char* ws = (char*)d_ws;
  bf16_t* adjp   = (bf16_t*)(ws + 0);
  bf16_t* wt1    = (bf16_t*)(ws + 194560);
  bf16_t* wt2    = (bf16_t*)(ws + 198656);
  bf16_t* h1     = (bf16_t*)(ws + 206848);
  bf16_t* h2p    = (bf16_t*)(ws + 59189248);
  bf16_t* cw1t   = (bf16_t*)(ws + 123086848);
  bf16_t* mw1t   = (bf16_t*)(ws + 137832448);
  float*  y1     = (float*)(ws + 157624320);
  float*  y2     = (float*)(ws + 158410752);
  bf16_t* pooled = (bf16_t*)(ws + 159197184);
  float*  z1     = (float*)(ws + 159213568);
  float*  z2     = (float*)(ws + 159344640);

  k_prep_adj  <<<(304 * 320 + 255) / 256, 256, 0, stream>>>(adj, adjp);
  k_prep_gcnw <<<(64 * 64 + 255) / 256, 256, 0, stream>>>(gw1, gw2, wt1, wt2);
  k_prep_convw<<<(C1_ * CONVK_ + 255) / 256, 256, 0, stream>>>(cw1, cw1t);
  k_prep_mlpw <<<(MLPIN_ * MH1_ + 255) / 256, 256, 0, stream>>>(mw1, mw1t);
  k_zero_pad  <<<(B_ * 2 * ND_ + 255) / 256, 256, 0, stream>>>(h2p);

  k_gcn<16, true ><<<B_ * T_, 256, 0, stream>>>(x,  wt1, adjp, gb1, h1,  T_,  0);
  k_gcn<64, false><<<B_ * T_, 256, 0, stream>>>(h1, wt2, adjp, gb2, h2p, TP_, 1);

  k_conv1<<<96, 256, 0, stream>>>(h2p, cw1t, cb1, y1);
  k_conv2<<<(B_ * C2_ * T_ + 255) / 256, 256, 0, stream>>>(y1, cw2, cb2, y2);
  k_pool <<<(B_ * C2_ + 255) / 256, 256, 0, stream>>>(y2, pooled);

  k_mlp1<<<16, 256, 0, stream>>>(h2p, pooled, mw1t, mb1, z1);
  k_mlp2<<<(B_ * MH2_ + 255) / 256, 256, 0, stream>>>(z1, mw2, mb2, z2);
  k_mlp3<<<(B_ * OUT_ + 255) / 256, 256, 0, stream>>>(z2, mw3, mb3, out);
}
